// FeedForward_52750788329630
// MI455X (gfx1250) — matrix operand order repaired and checked by reading, not compiled, not run on hardware
//
#include <hip/hip_runtime.h>
#include <hip/hip_bf16.h>

#define DIM 512
#define FFN 1408
#define NTOK 32768           // 8 * 4096 tokens
#define GROUP 128

typedef _Float16 v16h __attribute__((ext_vector_type(16)));
typedef _Float16 v8h  __attribute__((ext_vector_type(8)));
typedef _Float16 v4h  __attribute__((ext_vector_type(4)));
typedef float    v8f  __attribute__((ext_vector_type(8)));

// ---------------------------------------------------------------------------
// Kernel 1: convert x (f32) -> f16 workspace copy
// ---------------------------------------------------------------------------
__global__ void __launch_bounds__(256)
cvt_x_f16(const float* __restrict__ x, _Float16* __restrict__ xh, int n) {
    int i = (blockIdx.x * 256 + threadIdx.x) * 4;
    if (i >= n) return;
    float4 v = *(const float4*)(x + i);
    v4h o;
    o[0] = (_Float16)v.x; o[1] = (_Float16)v.y;
    o[2] = (_Float16)v.z; o[3] = (_Float16)v.w;
    *(v4h*)(xh + i) = o;
}

// ---------------------------------------------------------------------------
// Kernel 2: BitNet group-wise ternary quantization (scale baked into f16).
// One wave32 per (row, group-of-128): scale = mean|w| + eps over the group,
// wq = clip(round(w/scale),-1,1) * scale stored as f16 (2^-11 rel. error,
// same order as the x->f16 conversion).
// ---------------------------------------------------------------------------
__global__ void __launch_bounds__(256)
quant_weights(const float* __restrict__ w, _Float16* __restrict__ wq,
              int rows, int in_dim) {
    int wave = threadIdx.x >> 5;
    int lane = threadIdx.x & 31;
    int ngroups = in_dim >> 7;                 // in_dim / 128
    int gid = blockIdx.x * 8 + wave;           // (row, group) index
    int row = gid / ngroups;
    int grp = gid - row * ngroups;
    if (row >= rows) return;

    const float* base = w + row * in_dim + (grp << 7);
    float a0 = base[lane];
    float a1 = base[lane + 32];
    float a2 = base[lane + 64];
    float a3 = base[lane + 96];
    float s = fabsf(a0) + fabsf(a1) + fabsf(a2) + fabsf(a3);
    #pragma unroll
    for (int off = 16; off > 0; off >>= 1)     // wave32 butterfly reduce
        s += __shfl_xor(s, off, 32);
    float scale = s * (1.0f / 128.0f) + 1e-5f;
    float inv = 1.0f / scale;

    _Float16* qb = wq + row * in_dim + (grp << 7);
    qb[lane]      = (_Float16)(fminf(1.0f, fmaxf(-1.0f, rintf(a0 * inv))) * scale);
    qb[lane + 32] = (_Float16)(fminf(1.0f, fmaxf(-1.0f, rintf(a1 * inv))) * scale);
    qb[lane + 64] = (_Float16)(fminf(1.0f, fmaxf(-1.0f, rintf(a2 * inv))) * scale);
    qb[lane + 96] = (_Float16)(fminf(1.0f, fmaxf(-1.0f, rintf(a3 * inv))) * scale);
}

// ---------------------------------------------------------------------------
// WMMA helpers (CDNA5 16x16x32 f16, wave32)
// A 16x32 (k order chosen here, not the ISA table order): lane<16 -> M=lane,    K = 0..15 contiguous
//          lane>=16-> M=lane-16, K = 16..31 contiguous
// B 32x16 (same k order as A): lane<16 -> N=lane,    K = 0..15 contiguous
//          lane>=16-> N=lane-16, K = 16..31 contiguous
// C 16x16: lane L -> N = L&15, M = r + 8*(L>=16) for VGPR r
// ---------------------------------------------------------------------------
__device__ __forceinline__ v16h load_a_frag(const _Float16* p /* row + k0 + kb */) {
    v8h lo = *(const v8h*)(p);
    v8h hi = *(const v8h*)(p + 8);
    return __builtin_shufflevector(lo, hi, 0,1,2,3,4,5,6,7,8,9,10,11,12,13,14,15);
}

__device__ __forceinline__ v8f wmma32(v16h a, v16h b, v8f c) {
    return __builtin_amdgcn_wmma_f32_16x16x32_f16(false, a, false, b,
                                                  (short)0, c, false, false);
}

__device__ __forceinline__ void store_h_tile(_Float16* __restrict__ h,
                                             int mbase, int col, v8f g, v8f u) {
    _Float16* hp = h + mbase * FFN + col;
    #pragma unroll
    for (int r = 0; r < 8; ++r) {
        float gg = g[r];
        float uu = u[r];
        float sig = 1.0f / (1.0f + __expf(-gg));
        hp[r * FFN] = (_Float16)(gg * sig * uu);   // silu(gate) * up
    }
}

// ---------------------------------------------------------------------------
// Kernel 3: fused gate/up GEMM + SwiGLU -> h (f16)
// Each wave computes a 32Mx32N tile for BOTH gate and up: per k-step,
// 6 v16h loads feed 8 WMMAs (1.5 b128 loads per WMMA).
// Block = 8 waves arranged 2M x 4N -> block tile 64M x 128N.
// Grid: (FFN/128, NTOK/64).
// ---------------------------------------------------------------------------
__global__ void __launch_bounds__(256)
gateup_swiglu(const _Float16* __restrict__ xh,
              const _Float16* __restrict__ wqg,
              const _Float16* __restrict__ wqu,
              _Float16* __restrict__ h) {
    int lane  = threadIdx.x & 31;
    int wave  = threadIdx.x >> 5;
    int nwave = wave & 3;
    int mwave = wave >> 2;
    int n0 = blockIdx.x * 128 + nwave * 32;
    int m0 = blockIdx.y * 64  + mwave * 32;
    int l15  = lane & 15;
    int kb   = (lane & 16) ? 16 : 0;   // A K-chunk base
    int koff = (lane & 16) ? 16 : 0;   // B K-run base

    const _Float16* a0p = xh  + (m0 + l15) * DIM + kb;
    const _Float16* a1p = a0p + 16 * DIM;
    const _Float16* g0p = wqg + (n0 + l15) * DIM + koff;
    const _Float16* g1p = g0p + 16 * DIM;
    const _Float16* u0p = wqu + (n0 + l15) * DIM + koff;
    const _Float16* u1p = u0p + 16 * DIM;

    v8f cg00 = {}, cg01 = {}, cg10 = {}, cg11 = {};
    v8f cu00 = {}, cu01 = {}, cu10 = {}, cu11 = {};

    #pragma unroll 2
    for (int k0 = 0; k0 < DIM; k0 += 32) {
        v16h a0  = load_a_frag(a0p + k0);
        v16h a1  = load_a_frag(a1p + k0);
        v16h bg0 = *(const v16h*)(g0p + k0);
        v16h bg1 = *(const v16h*)(g1p + k0);
        v16h bu0 = *(const v16h*)(u0p + k0);
        v16h bu1 = *(const v16h*)(u1p + k0);
        cg00 = wmma32(a0, bg0, cg00);
        cg01 = wmma32(a0, bg1, cg01);
        cg10 = wmma32(a1, bg0, cg10);
        cg11 = wmma32(a1, bg1, cg11);
        cu00 = wmma32(a0, bu0, cu00);
        cu01 = wmma32(a0, bu1, cu01);
        cu10 = wmma32(a1, bu0, cu10);
        cu11 = wmma32(a1, bu1, cu11);
    }

    int mb0 = m0 + ((lane & 16) ? 8 : 0);
    int c0  = n0 + l15;
    store_h_tile(h, mb0,      c0,      cg00, cu00);
    store_h_tile(h, mb0,      c0 + 16, cg01, cu01);
    store_h_tile(h, mb0 + 16, c0,      cg10, cu10);
    store_h_tile(h, mb0 + 16, c0 + 16, cg11, cu11);
}

// ---------------------------------------------------------------------------
// Kernel 4: down projection GEMM: out = h @ wq_down^T (f32 out)
// Each wave computes 32Mx64N: 6 v16h loads per 8 WMMAs. A block's 8 waves
// span the full N=512 and share the same 32 A rows (L0 reuse).
// Grid: NTOK/32 blocks (1-D).
// ---------------------------------------------------------------------------
__global__ void __launch_bounds__(256)
down_proj(const _Float16* __restrict__ h,
          const _Float16* __restrict__ wqd,
          float* __restrict__ out) {
    int lane = threadIdx.x & 31;
    int wave = threadIdx.x >> 5;
    int n0 = wave * 64;                 // 8 waves * 64 = N = 512
    int m0 = blockIdx.x * 32;
    int l15  = lane & 15;
    int kb   = (lane & 16) ? 16 : 0;
    int koff = (lane & 16) ? 16 : 0;

    const _Float16* a0p = h + (m0 + l15) * FFN + kb;
    const _Float16* a1p = a0p + 16 * FFN;
    const _Float16* b0p = wqd + (n0 + l15) * FFN + koff;
    const _Float16* b1p = b0p + 16 * FFN;
    const _Float16* b2p = b0p + 32 * FFN;
    const _Float16* b3p = b0p + 48 * FFN;

    v8f c00 = {}, c01 = {}, c02 = {}, c03 = {};
    v8f c10 = {}, c11 = {}, c12 = {}, c13 = {};

    #pragma unroll 2
    for (int k0 = 0; k0 < FFN; k0 += 32) {
        v16h a0 = load_a_frag(a0p + k0);
        v16h a1 = load_a_frag(a1p + k0);
        v16h b0 = *(const v16h*)(b0p + k0);
        v16h b1 = *(const v16h*)(b1p + k0);
        v16h b2 = *(const v16h*)(b2p + k0);
        v16h b3 = *(const v16h*)(b3p + k0);
        c00 = wmma32(a0, b0, c00);
        c01 = wmma32(a0, b1, c01);
        c02 = wmma32(a0, b2, c02);
        c03 = wmma32(a0, b3, c03);
        c10 = wmma32(a1, b0, c10);
        c11 = wmma32(a1, b1, c11);
        c12 = wmma32(a1, b2, c12);
        c13 = wmma32(a1, b3, c13);
    }

    int mb0 = m0 + ((lane & 16) ? 8 : 0);
    int c0  = n0 + l15;
    v8f accs[8] = {c00, c01, c02, c03, c10, c11, c12, c13};
    #pragma unroll
    for (int t = 0; t < 8; ++t) {
        int mbase = mb0 + (t >> 2) * 16;
        int col   = c0 + (t & 3) * 16;
        float* op = out + mbase * DIM + col;
        #pragma unroll
        for (int r = 0; r < 8; ++r)
            op[r * DIM] = accs[t][r];
    }
}

// ---------------------------------------------------------------------------
// Host launcher
// ---------------------------------------------------------------------------
extern "C" void kernel_launch(void* const* d_in, const int* in_sizes, int n_in,
                              void* d_out, int out_size, void* d_ws, size_t ws_size,
                              hipStream_t stream) {
    const float* x      = (const float*)d_in[0];  // [8,4096,512]
    const float* w_gate = (const float*)d_in[1];  // [1408,512]
    const float* w_up   = (const float*)d_in[2];  // [1408,512]
    const float* w_down = (const float*)d_in[3];  // [512,1408]
    float* out = (float*)d_out;                   // [8,4096,512]

    // Workspace layout (all section sizes are multiples of 256 bytes)
    char* ws = (char*)d_ws;
    size_t off = 0;
    _Float16* xh   = (_Float16*)(ws + off); off += (size_t)NTOK * DIM * 2;   // 32 MiB
    _Float16* wqg  = (_Float16*)(ws + off); off += (size_t)FFN * DIM * 2;
    _Float16* wqu  = (_Float16*)(ws + off); off += (size_t)FFN * DIM * 2;
    _Float16* wqd  = (_Float16*)(ws + off); off += (size_t)DIM * FFN * 2;
    _Float16* hbuf = (_Float16*)(ws + off); off += (size_t)NTOK * FFN * 2;   // 88 MiB
    (void)ws_size; (void)in_sizes; (void)n_in; (void)out_size;

    // 1) x -> f16
    {
        int n = NTOK * DIM;                       // 16,777,216
        int blocks = n / (256 * 4);               // exact
        cvt_x_f16<<<blocks, 256, 0, stream>>>(x, xh, n);
    }
    // 2) quantize the three weight matrices (one wave per row-group)
    {
        int blocks_g = (FFN * (DIM / GROUP)) / 8; // 704
        quant_weights<<<blocks_g, 256, 0, stream>>>(w_gate, wqg, FFN, DIM);
        quant_weights<<<blocks_g, 256, 0, stream>>>(w_up,   wqu, FFN, DIM);
        int blocks_d = (DIM * (FFN / GROUP)) / 8; // 704
        quant_weights<<<blocks_d, 256, 0, stream>>>(w_down, wqd, DIM, FFN);
    }
    // 3) fused gate/up + SwiGLU -> h
    {
        dim3 grid(FFN / 128, NTOK / 64);          // (11, 512)
        gateup_swiglu<<<grid, 256, 0, stream>>>(xh, wqg, wqu, hbuf);
    }
    // 4) down projection -> out
    {
        down_proj<<<NTOK / 32, 256, 0, stream>>>(hbuf, wqd, out);
    }
}
